// GraphEncoderOneHead_40037685133632
// MI455X (gfx1250) — matrix operand order repaired and checked by reading, not compiled, not run on hardware
//
#include <hip/hip_runtime.h>
#include <hip/hip_bf16.h>

typedef __attribute__((ext_vector_type(16))) __bf16 v16bf;
typedef __attribute__((ext_vector_type(8)))  float  v8f;

#define DFEAT 64
#define EPSBN 1e-5f

// ---------------------------------------------------------------------------
// Generic utility kernels
// ---------------------------------------------------------------------------
__global__ void fill_kernel(float* __restrict__ p, float v, int n) {
  int i = blockIdx.x * blockDim.x + threadIdx.x;
  if (i < n) p[i] = v;
}

// ---------------------------------------------------------------------------
// WMMA bf16 hi/lo-split GEMM:  C = epilogue(A[MxK] @ B[KxN] + bias)
// One wave32 computes one 16x16 tile. K must be a multiple of 32,
// N a multiple of 16. Rows beyond M are clamped on load / masked on store.
// ---------------------------------------------------------------------------
__global__ void wmma_gemm_bf16split(const float* __restrict__ A, int lda,
                                    const float* __restrict__ B, int ldb,
                                    const float* __restrict__ bias,
                                    const float* __restrict__ scale_log, // out *= expf(*scale_log)
                                    float* __restrict__ C, int ldc,
                                    int M, int K, int Ncols, int do_relu) {
  const int lane  = threadIdx.x & 31;
  const int wave  = threadIdx.x >> 5;
  const int wpb   = blockDim.x >> 5;
  const int tile  = blockIdx.x * wpb + wave;
  const int tiles_n = Ncols >> 4;
  const int tiles_m = (M + 15) >> 4;
  if (tile >= tiles_m * tiles_n) return;   // wave-uniform exit: EXEC stays full
  const int tm = tile / tiles_n;
  const int tn = tile - tm * tiles_n;
  const int r0 = tm << 4, c0 = tn << 4;
  const int half = lane >> 4;      // 0: lanes 0-15, 1: lanes 16-31
  const int mr   = lane & 15;      // A-row / B-col / C-col within tile

  const int arow = min(r0 + mr, M - 1);    // clamp partial last tile (loads only)
  const int bcol = c0 + mr;

  v8f acc = {0.f, 0.f, 0.f, 0.f, 0.f, 0.f, 0.f, 0.f};

  for (int k0 = 0; k0 < K; k0 += 32) {
    v16bf ahi, alo, bhi, blo;
#pragma unroll
    for (int e = 0; e < 16; ++e) {
      // A 16-bit layout (ISA 7.12.2): VGPR j holds K = 16*(j>=4) + 8*half + 2*(j&3)+s
      int ka = ((e >> 3) << 4) + (half << 3) + (e & 7);
      float av = A[(size_t)arow * lda + (k0 + ka)];
      __bf16 ah = (__bf16)av;
      ahi[e] = ah;
      alo[e] = (__bf16)(av - (float)ah);
      // B 16-bit layout: same K order as A
      int kb = ((e >> 3) << 4) + (half << 3) + (e & 7);
      float bv = B[(size_t)(k0 + kb) * ldb + bcol];
      __bf16 bh = (__bf16)bv;
      bhi[e] = bh;
      blo[e] = (__bf16)(bv - (float)bh);
    }
    acc = __builtin_amdgcn_wmma_f32_16x16x32_bf16(false, ahi, false, bhi, (short)0, acc, false, false);
    acc = __builtin_amdgcn_wmma_f32_16x16x32_bf16(false, alo, false, bhi, (short)0, acc, false, false);
    acc = __builtin_amdgcn_wmma_f32_16x16x32_bf16(false, ahi, false, blo, (short)0, acc, false, false);
  }

  float sc = scale_log ? expf(*scale_log) : 1.0f;
  float bc = bias ? bias[bcol] : 0.0f;
#pragma unroll
  for (int v = 0; v < 8; ++v) {
    int r = r0 + v + (half << 3);          // C layout: VGPR v -> row v (+8 for upper half)
    if (r < M) {
      float val = acc[v] + bc;
      if (do_relu) val = fmaxf(val, 0.0f);
      C[(size_t)r * ldc + bcol] = val * sc;
    }
  }
}

// ---------------------------------------------------------------------------
// GCN: degree / normalization
// ---------------------------------------------------------------------------
__global__ void deg_kernel(const long long* __restrict__ dst, float* __restrict__ deg, int E) {
  int e = blockIdx.x * blockDim.x + threadIdx.x;
  if (e < E) atomicAdd(&deg[(int)dst[e]], 1.0f);
}
__global__ void dinv_kernel(const float* __restrict__ deg, float* __restrict__ dinv, int n) {
  int i = blockIdx.x * blockDim.x + threadIdx.x;
  if (i < n) dinv[i] = rsqrtf(deg[i]);
}

// agg[i,c] = h[i,c]*dinv[i]^2 + b[c]   (self-loop term + bias)
__global__ void gcn_self_init(const float* __restrict__ h, const float* __restrict__ dinv,
                              const float* __restrict__ b, float* __restrict__ agg, int n) {
  int idx = blockIdx.x * blockDim.x + threadIdx.x;
  if (idx >= n * DFEAT) return;
  int i = idx >> 6, c = idx & 63;
  float di = dinv[i];
  agg[idx] = h[idx] * di * di + b[c];
}

// 16 threads per edge; each handles 4 features via float4 gather + 4 atomics
__global__ void gcn_scatter(const long long* __restrict__ src, const long long* __restrict__ dst,
                            const float* __restrict__ h, const float* __restrict__ dinv,
                            float* __restrict__ agg, int E) {
  int t = blockIdx.x * blockDim.x + threadIdx.x;
  if (t >= E * 16) return;
  int e = t >> 4, q = t & 15;
  int s = (int)src[e], d = (int)dst[e];
  float w = dinv[s] * dinv[d];
  float4 hv = *reinterpret_cast<const float4*>(h + (size_t)s * DFEAT + q * 4);
  float* ap = agg + (size_t)d * DFEAT + q * 4;
  atomicAdd(ap + 0, hv.x * w);
  atomicAdd(ap + 1, hv.y * w);
  atomicAdd(ap + 2, hv.z * w);
  atomicAdd(ap + 3, hv.w * w);
}

// ---------------------------------------------------------------------------
// BatchNorm (training-mode stats over nodes, biased variance)
// ---------------------------------------------------------------------------
__global__ void bn_stats(const float* __restrict__ v, float* __restrict__ s1,
                         float* __restrict__ s2, int n) {
  __shared__ float ls1[DFEAT], ls2[DFEAT];
  int c = threadIdx.x & 63;
  int rowsPerBlk = blockDim.x >> 6;
  int sub = threadIdx.x >> 6;
  if (threadIdx.x < DFEAT) { ls1[threadIdx.x] = 0.f; ls2[threadIdx.x] = 0.f; }
  __syncthreads();
  float a = 0.f, b = 0.f;
  for (int r = blockIdx.x * rowsPerBlk + sub; r < n; r += gridDim.x * rowsPerBlk) {
    float x = v[(size_t)r * DFEAT + c];
    a += x; b += x * x;
  }
  atomicAdd(&ls1[c], a);
  atomicAdd(&ls2[c], b);
  __syncthreads();
  if (threadIdx.x < DFEAT) {
    atomicAdd(&s1[threadIdx.x], ls1[threadIdx.x]);
    atomicAdd(&s2[threadIdx.x], ls2[threadIdx.x]);
  }
}

// x = relu(gamma*(v-mu)*rsqrt(var+eps)+beta) + x   (in-place residual)
__global__ void bn_apply_res(const float* __restrict__ v, const float* __restrict__ g,
                             const float* __restrict__ be, const float* __restrict__ s1,
                             const float* __restrict__ s2, float* __restrict__ x, int n) {
  int idx = blockIdx.x * blockDim.x + threadIdx.x;
  if (idx >= n * DFEAT) return;
  int c = idx & 63;
  float invn = 1.0f / (float)n;
  float mu = s1[c] * invn;
  float var = s2[c] * invn - mu * mu;
  float y = g[c] * (v[idx] - mu) * rsqrtf(var + EPSBN) + be[c];
  x[idx] = fmaxf(y, 0.0f) + x[idx];
}

// ---------------------------------------------------------------------------
// GAT (heads=1, self loops, LeakyReLU(0.2), edge softmax)
// ---------------------------------------------------------------------------
__device__ __forceinline__ float lrelu02(float x) { return x > 0.f ? x : 0.2f * x; }
__device__ __forceinline__ unsigned fenc(float f) {
  unsigned u = __float_as_uint(f);
  return (u & 0x80000000u) ? ~u : (u | 0x80000000u);
}
__device__ __forceinline__ float fdec(unsigned u) {
  u = (u & 0x80000000u) ? (u & 0x7fffffffu) : ~u;
  return __uint_as_float(u);
}

__global__ void gat_scores(const float* __restrict__ h, const float* __restrict__ a_src,
                           const float* __restrict__ a_dst, float* __restrict__ ssrc,
                           float* __restrict__ sdst, unsigned* __restrict__ mbits, int n) {
  int i = blockIdx.x * blockDim.x + threadIdx.x;
  if (i >= n) return;
  const float* hr = h + (size_t)i * DFEAT;
  float s1 = 0.f, s2 = 0.f;
#pragma unroll
  for (int c = 0; c < DFEAT; c += 4) {
    float4 hv = *reinterpret_cast<const float4*>(hr + c);
    float4 as = *reinterpret_cast<const float4*>(a_src + c);
    float4 ad = *reinterpret_cast<const float4*>(a_dst + c);
    s1 += hv.x * as.x + hv.y * as.y + hv.z * as.z + hv.w * as.w;
    s2 += hv.x * ad.x + hv.y * ad.y + hv.z * ad.z + hv.w * ad.w;
  }
  ssrc[i] = s1;
  sdst[i] = s2;
  mbits[i] = fenc(lrelu02(s1 + s2));   // self-loop initializes the segment max
}

__global__ void gat_max(const long long* __restrict__ src, const long long* __restrict__ dst,
                        const float* __restrict__ ssrc, const float* __restrict__ sdst,
                        unsigned* __restrict__ mbits, int E) {
  int e = blockIdx.x * blockDim.x + threadIdx.x;
  if (e >= E) return;
  int s = (int)src[e], d = (int)dst[e];
  atomicMax(&mbits[d], fenc(lrelu02(ssrc[s] + sdst[d])));
}

__global__ void gat_denom_init(const unsigned* __restrict__ mbits, const float* __restrict__ ssrc,
                               const float* __restrict__ sdst, float* __restrict__ mbuf,
                               float* __restrict__ denom, int n) {
  int i = blockIdx.x * blockDim.x + threadIdx.x;
  if (i >= n) return;
  float m = fdec(mbits[i]);
  mbuf[i] = m;
  denom[i] = expf(lrelu02(ssrc[i] + sdst[i]) - m);  // self-loop term
}

__global__ void gat_denom_edges(const long long* __restrict__ src, const long long* __restrict__ dst,
                                const float* __restrict__ ssrc, const float* __restrict__ sdst,
                                const float* __restrict__ mbuf, float* __restrict__ denom, int E) {
  int e = blockIdx.x * blockDim.x + threadIdx.x;
  if (e >= E) return;
  int s = (int)src[e], d = (int)dst[e];
  atomicAdd(&denom[d], expf(lrelu02(ssrc[s] + sdst[d]) - mbuf[d]));
}

// agg[i,c] = alpha_self * h[i,c] + bg[c]
__global__ void gat_self_out(const float* __restrict__ h, const float* __restrict__ ssrc,
                             const float* __restrict__ sdst, const float* __restrict__ mbuf,
                             const float* __restrict__ denom, const float* __restrict__ bg,
                             float* __restrict__ agg, int n) {
  int idx = blockIdx.x * blockDim.x + threadIdx.x;
  if (idx >= n * DFEAT) return;
  int i = idx >> 6, c = idx & 63;
  float coef = expf(lrelu02(ssrc[i] + sdst[i]) - mbuf[i]) / denom[i];
  agg[idx] = coef * h[idx] + bg[c];
}

__global__ void gat_scatter(const long long* __restrict__ src, const long long* __restrict__ dst,
                            const float* __restrict__ h, const float* __restrict__ ssrc,
                            const float* __restrict__ sdst, const float* __restrict__ mbuf,
                            const float* __restrict__ denom, float* __restrict__ agg, int E) {
  int t = blockIdx.x * blockDim.x + threadIdx.x;
  if (t >= E * 16) return;
  int e = t >> 4, q = t & 15;
  int s = (int)src[e], d = (int)dst[e];
  float coef = expf(lrelu02(ssrc[s] + sdst[d]) - mbuf[d]) / denom[d];
  float4 hv = *reinterpret_cast<const float4*>(h + (size_t)s * DFEAT + q * 4);
  float* ap = agg + (size_t)d * DFEAT + q * 4;
  atomicAdd(ap + 0, hv.x * coef);
  atomicAdd(ap + 1, hv.y * coef);
  atomicAdd(ap + 2, hv.z * coef);
  atomicAdd(ap + 3, hv.w * coef);
}

// ---------------------------------------------------------------------------
// Global mean pool
// ---------------------------------------------------------------------------
__global__ void cnt_kernel(const long long* __restrict__ batch, float* __restrict__ cnt, int n) {
  int i = blockIdx.x * blockDim.x + threadIdx.x;
  if (i < n) atomicAdd(&cnt[(int)batch[i]], 1.0f);
}
__global__ void pool_kernel(const long long* __restrict__ batch, const float* __restrict__ x,
                            float* __restrict__ pooled, int n) {
  int idx = blockIdx.x * blockDim.x + threadIdx.x;
  if (idx >= n * DFEAT) return;
  int i = idx >> 6, c = idx & 63;
  atomicAdd(&pooled[(size_t)batch[i] * DFEAT + c], x[idx]);
}
__global__ void pool_div(float* __restrict__ pooled, const float* __restrict__ cnt, int g) {
  int idx = blockIdx.x * blockDim.x + threadIdx.x;
  if (idx >= g * DFEAT) return;
  pooled[idx] /= fmaxf(cnt[idx >> 6], 1.0f);
}

// ---------------------------------------------------------------------------
// Host-side orchestration
// ---------------------------------------------------------------------------
static inline int ceil_div(long long a, int b) { return (int)((a + b - 1) / b); }

static void launch_gemm(const float* A, int lda, const float* B, int ldb,
                        const float* bias, const float* scale_log,
                        float* C, int ldc, int M, int K, int Ncols, int relu,
                        hipStream_t stream) {
  long long tiles = (long long)((M + 15) / 16) * (Ncols / 16);
  dim3 grid(ceil_div(tiles, 4)), block(128);   // 4 waves / block
  hipLaunchKernelGGL(wmma_gemm_bf16split, grid, block, 0, stream,
                     A, lda, B, ldb, bias, scale_log, C, ldc, M, K, Ncols, relu);
}

extern "C" void kernel_launch(void* const* d_in, const int* in_sizes, int n_in,
                              void* d_out, int out_size, void* d_ws, size_t ws_size,
                              hipStream_t stream) {
  const int NHID = 256, NOUT = 256;
  const int N = in_sizes[0] / DFEAT;
  const int E = in_sizes[1] / 2;
  const int G = out_size / NOUT;

  const float*     x_in  = (const float*)d_in[0];
  const long long* ei    = (const long long*)d_in[1];     // int64 [2,E]
  const long long* src   = ei;
  const long long* dst   = ei + E;
  const long long* batch = (const long long*)d_in[2];     // int64 [N]

  const float* W[3]  = {(const float*)d_in[3], (const float*)d_in[5], (const float*)d_in[7]};
  const float* b[3]  = {(const float*)d_in[4], (const float*)d_in[6], (const float*)d_in[8]};
  const float* g[3]  = {(const float*)d_in[9], (const float*)d_in[11], (const float*)d_in[13]};
  const float* be[3] = {(const float*)d_in[10], (const float*)d_in[12], (const float*)d_in[14]};
  const float* ga  = (const float*)d_in[15];
  const float* bea = (const float*)d_in[16];
  const float* Wg  = (const float*)d_in[17];
  const float* a_s = (const float*)d_in[18];
  const float* a_d = (const float*)d_in[19];
  const float* bg  = (const float*)d_in[20];
  const float* Wh1 = (const float*)d_in[21];
  const float* bh1 = (const float*)d_in[22];
  const float* Wm0 = (const float*)d_in[23];
  const float* bm0 = (const float*)d_in[24];
  const float* Wh2 = (const float*)d_in[25];
  const float* bh2 = (const float*)d_in[26];
  const float* temp = (const float*)d_in[27];

  // Workspace layout (all fp32)
  float* w = (float*)d_ws;
  float*    xcur  = w; w += (size_t)N * DFEAT;
  float*    hbuf  = w; w += (size_t)N * DFEAT;
  float*    agg   = w; w += (size_t)N * DFEAT;
  float*    deg   = w; w += N;
  float*    dinv  = w; w += N;
  float*    ssrc  = w; w += N;
  float*    sdst  = w; w += N;
  float*    mbuf  = w; w += N;
  unsigned* mbits = (unsigned*)w; w += N;
  float*    denom = w; w += N;
  float*    colsum = w; w += DFEAT;    // colsum + colsq contiguous (zeroed together)
  float*    colsq  = w; w += DFEAT;
  float*    cnt    = w; w += G;
  float*    pooled = w; w += (size_t)G * DFEAT;
  float*    h1     = w; w += (size_t)G * NHID;
  float*    h2     = w; w += (size_t)G * NHID;
  (void)ws_size; (void)n_in;

  const int BS = 256;
  const int gN  = ceil_div(N, BS);
  const int gND = ceil_div((long long)N * DFEAT, BS);
  const int gE  = ceil_div(E, BS);
  const int gE16 = ceil_div((long long)E * 16, BS);

  // x working copy (inputs must not be mutated)
  hipMemcpyAsync(xcur, x_in, (size_t)N * DFEAT * sizeof(float),
                 hipMemcpyDeviceToDevice, stream);

  // Degree normalization (shared by all 3 GCN layers): deg = 1 + indegree
  hipLaunchKernelGGL(fill_kernel, dim3(gN), dim3(BS), 0, stream, deg, 1.0f, N);
  hipLaunchKernelGGL(deg_kernel, dim3(gE), dim3(BS), 0, stream, dst, deg, E);
  hipLaunchKernelGGL(dinv_kernel, dim3(gN), dim3(BS), 0, stream, deg, dinv, N);

  // ---- 3 GCN blocks ----
  for (int l = 0; l < 3; ++l) {
    launch_gemm(xcur, DFEAT, W[l], DFEAT, nullptr, nullptr, hbuf, DFEAT,
                N, DFEAT, DFEAT, 0, stream);                           // h = x @ W  (WMMA)
    hipLaunchKernelGGL(gcn_self_init, dim3(gND), dim3(BS), 0, stream, hbuf, dinv, b[l], agg, N);
    hipLaunchKernelGGL(gcn_scatter, dim3(gE16), dim3(BS), 0, stream, src, dst, hbuf, dinv, agg, E);
    hipLaunchKernelGGL(fill_kernel, dim3(1), dim3(128), 0, stream, colsum, 0.0f, 2 * DFEAT);
    hipLaunchKernelGGL(bn_stats, dim3(512), dim3(BS), 0, stream, agg, colsum, colsq, N);
    hipLaunchKernelGGL(bn_apply_res, dim3(gND), dim3(BS), 0, stream, agg, g[l], be[l],
                       colsum, colsq, xcur, N);
  }

  // ---- GAT block ----
  launch_gemm(xcur, DFEAT, Wg, DFEAT, nullptr, nullptr, hbuf, DFEAT,
              N, DFEAT, DFEAT, 0, stream);                             // h = x @ Wg (WMMA)
  hipLaunchKernelGGL(gat_scores, dim3(gN), dim3(BS), 0, stream, hbuf, a_s, a_d, ssrc, sdst, mbits, N);
  hipLaunchKernelGGL(gat_max, dim3(gE), dim3(BS), 0, stream, src, dst, ssrc, sdst, mbits, E);
  hipLaunchKernelGGL(gat_denom_init, dim3(gN), dim3(BS), 0, stream, mbits, ssrc, sdst, mbuf, denom, N);
  hipLaunchKernelGGL(gat_denom_edges, dim3(gE), dim3(BS), 0, stream, src, dst, ssrc, sdst, mbuf, denom, E);
  hipLaunchKernelGGL(gat_self_out, dim3(gND), dim3(BS), 0, stream, hbuf, ssrc, sdst, mbuf, denom, bg, agg, N);
  hipLaunchKernelGGL(gat_scatter, dim3(gE16), dim3(BS), 0, stream, src, dst, hbuf, ssrc, sdst, mbuf, denom, agg, E);
  hipLaunchKernelGGL(fill_kernel, dim3(1), dim3(128), 0, stream, colsum, 0.0f, 2 * DFEAT);
  hipLaunchKernelGGL(bn_stats, dim3(512), dim3(BS), 0, stream, agg, colsum, colsq, N);
  hipLaunchKernelGGL(bn_apply_res, dim3(gND), dim3(BS), 0, stream, agg, ga, bea, colsum, colsq, xcur, N);

  // ---- global mean pool ----
  hipLaunchKernelGGL(fill_kernel, dim3(ceil_div(G, BS)), dim3(BS), 0, stream, cnt, 0.0f, G);
  hipLaunchKernelGGL(fill_kernel, dim3(ceil_div((long long)G * DFEAT, BS)), dim3(BS), 0, stream,
                     pooled, 0.0f, G * DFEAT);
  hipLaunchKernelGGL(cnt_kernel, dim3(gN), dim3(BS), 0, stream, batch, cnt, N);
  hipLaunchKernelGGL(pool_kernel, dim3(gND), dim3(BS), 0, stream, batch, xcur, pooled, N);
  hipLaunchKernelGGL(pool_div, dim3(ceil_div((long long)G * DFEAT, BS)), dim3(BS), 0, stream,
                     pooled, cnt, G);

  // ---- MLP head (all WMMA) ----
  launch_gemm(pooled, DFEAT, Wh1, NHID, bh1, nullptr, h1, NHID, G, DFEAT, NHID, 1, stream);
  launch_gemm(h1, NHID, Wm0, NHID, bm0, nullptr, h2, NHID, G, NHID, NHID, 1, stream);
  launch_gemm(h2, NHID, Wh2, NOUT, bh2, temp, (float*)d_out, NOUT, G, NHID, NOUT, 0, stream);
}
